// DynamicWeightProjection_12756052869653
// MI455X (gfx1250) — matrix operand order repaired and checked by reading, not compiled, not run on hardware
//
#include <hip/hip_runtime.h>
#include <hip/hip_bf16.h>
#include <math.h>

typedef __attribute__((ext_vector_type(16))) __bf16 v16bf;
typedef __attribute__((ext_vector_type(8)))  float  v8f;

// ---- problem dims ----
#define BT   16384
#define DK   4096
#define NCOL 640     // 512 (gelu->H) + 128 (tanh->dd)

// ---- output float offsets (tuple flattened in return order) ----
#define OFF_PRE_QW1   0u
#define OFF_PRE_QW2   1048576u
#define OFF_PRE_KW1   2097152u
#define OFF_PRE_KW2   3145728u
#define OFF_PRE_QDD   4194304u
#define OFF_PRE_KDD   4718592u
#define OFF_POST_QW1  5242880u
#define OFF_POST_QW2  6291456u
#define OFF_POST_KW1  7340032u
#define OFF_POST_KW2  8388608u
#define OFF_POST_QDD  9437184u
#define OFF_POST_KDD  9961472u
#define OFF_KW        10485760u

static __device__ __forceinline__ unsigned f2bf_bits(float f) {
    unsigned u = __float_as_uint(f);
    return ((u + 0x7FFFu + ((u >> 16) & 1u)) >> 16) & 0xFFFFu;   // RNE
}
static __device__ __forceinline__ unsigned pack_bf16(float a, float b) {
    return f2bf_bits(a) | (f2bf_bits(b) << 16);
}

// ============================================================================
// Prep: bf16-convert weights into WMMA-fragment-friendly packed dwords.
//  wpk : [2048][640]  dword(p,n) = {Wcat[2p][n], Wcat[2p+1][n]}  (bf16 lo/hi)
//        Wcat[d][n] = n<512 ? dw1[d*512+n] : dd_w[d*128+n-512]
//  qk2 : fragment-major: index ((c*4+ks)*128 + n)*16 + kpl
//        holds {qkw[c][k][n], qkw[c][k+1][n]} with k = ks*32 + 2*kpl
//        -> a B fragment is two 4-dword runs per lane, kpl = half*4 and 8 + half*4 (two b128 loads)
// ============================================================================
__global__ __launch_bounds__(256) void kPrep(const float* __restrict__ dw1,
                                             const float* __restrict__ qkw,
                                             const float* __restrict__ dd_w,
                                             unsigned* __restrict__ wpk,
                                             unsigned* __restrict__ qk2) {
    const int idx = blockIdx.x * 256 + threadIdx.x;
    if (idx < 2048 * 640) {
        const int p = idx / 640, n = idx % 640;
        const int d0 = 2 * p;
        float a, b;
        if (n < 512) {
            a = dw1[(size_t)d0 * 512 + n];
            b = dw1[(size_t)(d0 + 1) * 512 + n];
        } else {
            a = dd_w[(size_t)d0 * 128 + (n - 512)];
            b = dd_w[(size_t)(d0 + 1) * 128 + (n - 512)];
        }
        wpk[idx] = pack_bf16(a, b);
    } else {
        const int e = idx - 2048 * 640;
        if (e < 4 * 64 * 128) {
            const int c = e / 8192, r = e % 8192, kp = r / 128, n = r % 128;
            const int i = n >> 5, m = n & 31;
            const float a = qkw[(((size_t)c * 128 + 2 * kp    ) * 4 + i) * 32 + m];
            const float b = qkw[(((size_t)c * 128 + 2 * kp + 1) * 4 + i) * 32 + m];
            const int ks = kp >> 4, kpl = kp & 15;
            qk2[((size_t)(c * 4 + ks) * 128 + n) * 16 + kpl] = pack_bf16(a, b);
        }
    }
}

// ============================================================================
// Kernel A: [BT x 4096] x [4096 x 640] GEMM (bf16 WMMA, f32 accum).
// Block: 32 token rows x all 640 cols (Q read from HBM exactly once).
// 8 waves: rg = wave&1 (16-row group), cg = wave>>1 (160-col group, 10 tiles).
// LDS layouts are fragment-major so every WMMA operand is 2x ds_load_b128.
// ============================================================================
__global__ __launch_bounds__(256) void kA(const float* __restrict__ q,
                                          const unsigned* __restrict__ wpk,
                                          unsigned short* __restrict__ hOut, // bf16 [BT][512]
                                          float* __restrict__ out) {
    __shared__ unsigned qs[32 * 20];     // [row m][16 kpair], stride 20 (b128-aligned, conflict-free)
    __shared__ unsigned wsp[640 * 20];   // [col n][16 kpair], stride 20 (transposed!)

    const int tid  = threadIdx.x;
    const int lane = tid & 31;
    const int wv   = tid >> 5;
    const int rg   = wv & 1;
    const int cg   = wv >> 1;
    const int half = lane >> 4;
    const int l16  = lane & 15;
    const int rowBase = blockIdx.x * 32;

    v8f acc[10];
    const v8f vzero = {0.f, 0.f, 0.f, 0.f, 0.f, 0.f, 0.f, 0.f};
#pragma unroll
    for (int j = 0; j < 10; ++j) acc[j] = vzero;

    const int qrow = tid >> 3;        // 0..31
    const int qcol = (tid & 7) * 4;   // 0..28, step 4 within 32-wide K chunk

    for (int kb = 0; kb < DK; kb += 32) {
        // --- stage Q 32x32 fp32 -> bf16 kpair dwords, [m][kp] ---
        {
            const float4 qv = *(const float4*)(q + (size_t)(rowBase + qrow) * DK + kb + qcol);
            const int kp0 = qcol >> 1;
            qs[qrow * 20 + kp0]     = pack_bf16(qv.x, qv.y);
            qs[qrow * 20 + kp0 + 1] = pack_bf16(qv.z, qv.w);
        }
        // --- stage W chunk transposed: wsp[n][kp] <- wpk[p0+kp][n] ---
        {
            const int p0 = kb >> 1;
#pragma unroll
            for (int i = 0; i < 10; ++i) {
                const int idx4 = tid + i * 256;          // uint4 index, 0..2559
                const int r  = idx4 / 160;               // kpair row 0..15
                const int c4 = idx4 % 160;               // col group (4 cols)
                const uint4 w4 = *(const uint4*)(wpk + (size_t)(p0 + r) * 640 + c4 * 4);
                wsp[(c4 * 4 + 0) * 20 + r] = w4.x;
                wsp[(c4 * 4 + 1) * 20 + r] = w4.y;
                wsp[(c4 * 4 + 2) * 20 + r] = w4.z;
                wsp[(c4 * 4 + 3) * 20 + r] = w4.w;
            }
        }
        // --- prefetch next K-chunk (global_prefetch_b8) ---
        if (kb + 32 < DK) {
            __builtin_prefetch(q + (size_t)(rowBase + qrow) * DK + kb + 32 + qcol, 0, 1);
            __builtin_prefetch(wpk + (size_t)((kb >> 1) + 16 + (tid >> 4)) * 640 + (tid & 15) * 40, 0, 1);
        }
        __syncthreads();

        // --- A fragment: two contiguous b128 LDS loads per lane ---
        union { uint4 q2[2]; v16bf v; } af;
        {
            const int m = rg * 16 + l16;
            af.q2[0] = *(const uint4*)&qs[m * 20 + half * 4];       // v0..3: kp = half*4 + 0..3
            af.q2[1] = *(const uint4*)&qs[m * 20 + 8 + half * 4];   // v4..7: kp = 8 + half*4 + 0..3
        }
        // --- 10 N-tiles: B fragment (two b128 loads) + WMMA ---
#pragma unroll
        for (int j = 0; j < 10; ++j) {
            const int n = cg * 160 + j * 16 + l16;
            union { uint4 q2[2]; v16bf v; } bfr;
            bfr.q2[0] = *(const uint4*)&wsp[n * 20 + half * 4];     // kp = half*4 + 0..3
            bfr.q2[1] = *(const uint4*)&wsp[n * 20 + 8 + half * 4]; // kp = 8 + half*4 + 0..3
            acc[j] = __builtin_amdgcn_wmma_f32_16x16x32_bf16(
                false, af.v, false, bfr.v, (short)0, acc[j], false, false);
        }
        __syncthreads();
    }

    // --- epilogue: gelu(exact erf) -> H bf16 ; tanh -> dd outputs ---
    const unsigned ddOff[4] = {OFF_PRE_QDD, OFF_PRE_KDD, OFF_POST_QDD, OFF_POST_KDD};
#pragma unroll
    for (int j = 0; j < 10; ++j) {
        const int n = cg * 160 + j * 16 + l16;
#pragma unroll
        for (int v = 0; v < 8; ++v) {
            const int t = rowBase + rg * 16 + half * 8 + v;
            const float x = acc[j][v];
            if (n < 512) {
                const float g = 0.5f * x * (1.0f + erff(x * 0.70710678118654752f));
                hOut[(size_t)t * 512 + n] = (unsigned short)f2bf_bits(g);
            } else {
                const int dcol = n - 512;
                out[ddOff[dcol >> 5] + (size_t)t * 32 + (dcol & 31)] = tanhf(x);
            }
        }
    }
}

// ============================================================================
// Kernel B: per c: [16 tok x 128] x [128 x 128] bf16 WMMA; then rmsnorm(w1),
// write qw/kw outputs, and KW = kw1n^T kw2 + diag(kdd) via lane shuffles.
// Block = 128 threads (4 waves), each wave owns 16 tokens; no cross-wave sync.
// A/B fragments are explicit uint4 global loads (b128, cache-resident).
// ============================================================================
__global__ __launch_bounds__(128) void kB(const unsigned* __restrict__ hD,  // H dwords [BT][256]
                                          const unsigned* __restrict__ qk2, // frag-major packed qkw
                                          float* __restrict__ out) {
    __shared__ float wst[4][16 * 132];   // per-wave stage [16 tok][128 n], pad 132

    const int tid  = threadIdx.x;
    const int lane = tid & 31;
    const int wv   = tid >> 5;
    const int half = lane >> 4;
    const int l16  = lane & 15;
    const int tokBase = blockIdx.x * 64 + wv * 16;

    const unsigned w1Off[4]   = {OFF_PRE_QW1, OFF_PRE_KW1, OFF_POST_QW1, OFF_POST_KW1};
    const unsigned w2Off[4]   = {OFF_PRE_QW2, OFF_PRE_KW2, OFF_POST_QW2, OFF_POST_KW2};
    const unsigned kddOff2[2] = {OFF_PRE_KDD, OFF_POST_KDD};
    const v8f vzero = {0.f, 0.f, 0.f, 0.f, 0.f, 0.f, 0.f, 0.f};

    for (int c = 0; c < 4; ++c) {
        v8f acc[8];
#pragma unroll
        for (int j = 0; j < 8; ++j) acc[j] = vzero;

#pragma unroll
        for (int ks = 0; ks < 4; ++ks) {          // K = 128 = 4 x 32
            union { uint4 q2[2]; v16bf v; } af;
            const unsigned abase = (unsigned)(tokBase + l16) * 256 + c * 64 + ks * 16;
            af.q2[0] = *(const uint4*)(hD + abase + half * 4);      // kp = half*4 + 0..3
            af.q2[1] = *(const uint4*)(hD + abase + 8 + half * 4);  // kp = 8 + half*4 + 0..3
#pragma unroll
            for (int j = 0; j < 8; ++j) {
                union { uint4 q2[2]; v16bf v; } bfr;
                const unsigned bbase = ((unsigned)(c * 4 + ks) * 128 + j * 16 + l16) * 16;
                bfr.q2[0] = *(const uint4*)(qk2 + bbase + half * 4);
                bfr.q2[1] = *(const uint4*)(qk2 + bbase + 8 + half * 4);
                acc[j] = __builtin_amdgcn_wmma_f32_16x16x32_bf16(
                    false, af.v, false, bfr.v, (short)0, acc[j], false, false);
            }
        }

        // stage D tiles -> LDS (wave-private; compiler inserts ds waits)
#pragma unroll
        for (int j = 0; j < 8; ++j) {
#pragma unroll
            for (int v = 0; v < 8; ++v) {
                wst[wv][(half * 8 + v) * 132 + j * 16 + l16] = acc[j][v];
            }
        }

        const bool isK = (c & 1);
        const int  sel = c >> 1;                  // c==1 -> 0 (pre), c==3 -> 1 (post)
        for (int tk = 0; tk < 16; ++tk) {
            const int t = tokBase + tk;
            const float w0  = wst[wv][tk * 132 +  0 + lane];
            const float w1v = wst[wv][tk * 132 + 32 + lane];
            const float w2a = wst[wv][tk * 132 + 64 + lane];
            const float w2b = wst[wv][tk * 132 + 96 + lane];
            float s0 = w0 * w0, s1 = w1v * w1v;
#pragma unroll
            for (int off = 16; off > 0; off >>= 1) {
                s0 += __shfl_xor(s0, off, 32);
                s1 += __shfl_xor(s1, off, 32);
            }
            const float r0  = rsqrtf(s0 * (1.f / 32.f) + 1e-6f);
            const float r1  = rsqrtf(s1 * (1.f / 32.f) + 1e-6f);
            const float w0n = w0 * r0, w1n = w1v * r1;

            out[w1Off[c] + (size_t)t * 64 +      lane] = w0n;
            out[w1Off[c] + (size_t)t * 64 + 32 + lane] = w1n;
            out[w2Off[c] + (size_t)t * 64 +      lane] = w2a;
            out[w2Off[c] + (size_t)t * 64 + 32 + lane] = w2b;

            if (isK) {
                const float kddv = out[kddOff2[sel] + (size_t)t * 32 + lane];
                for (int m = 0; m < 32; ++m) {
                    const float a0 = __shfl(w0n, m, 32);
                    const float a1 = __shfl(w1n, m, 32);
                    const float km = __shfl(kddv, m, 32);
                    const float val = a0 * w2a + a1 * w2b + ((lane == m) ? km : 0.f);
                    out[OFF_KW + (size_t)t * 2048 + (size_t)sel * 1024 + m * 32 + lane] = val;
                }
            }
        }
    }
}

extern "C" void kernel_launch(void* const* d_in, const int* in_sizes, int n_in,
                              void* d_out, int out_size, void* d_ws, size_t ws_size,
                              hipStream_t stream) {
    const float* q    = (const float*)d_in[0];   // [4,4096,4096]
    const float* dw1  = (const float*)d_in[1];   // [4096,1,4,128]
    const float* qkw  = (const float*)d_in[2];   // [1,4,128,4,32]
    const float* dd_w = (const float*)d_in[3];   // [4096,1,128]
    float* out = (float*)d_out;

    char* ws = (char*)d_ws;
    unsigned*       wpk = (unsigned*)ws;                              // 5,242,880 B
    unsigned*       qk2 = (unsigned*)(ws + 5242880);                  //   131,072 B
    unsigned short* H   = (unsigned short*)(ws + 5242880 + 131072);   // 16,777,216 B

    // 2048*640 + 4*64*128 = 1,343,488 elements -> 5248 blocks of 256
    kPrep<<<5248, 256, 0, stream>>>(dw1, qkw, dd_w, wpk, qk2);
    kA<<<BT / 32, 256, 0, stream>>>(q, wpk, H, out);
    kB<<<BT / 64, 128, 0, stream>>>((const unsigned*)H, qk2, out);
}
